// MoE_32177894981790
// MI455X (gfx1250) — matrix operand order repaired and checked by reading, not compiled, not run on hardware
//
#include <hip/hip_runtime.h>
#include <hip/hip_bf16.h>
#include <stdint.h>

// ---------------------------------------------------------------------------
// MoE top-2 (T=8192, D=1024, H=2048, E=8) for gfx1250 (MI455X, wave32, WMMA).
// Sparse-routed, fused 2-layer expert MLP using v_wmma_f32_16x16x32_bf16
// with f32 accumulation. Weights/activations pre-converted to bf16 (K-major
// for B operands) in workspace. Workspace need: ~84.5 MB.
// Token tile M=32 doubles flop/byte vs M=16 (L2 weight traffic ~4.1 GB).
// ---------------------------------------------------------------------------

#define DIM    1024
#define INTER  2048
#define NEXP   8
#define NTOK   8192
#define MTOK   32            // tokens per tile
#define XS     (DIM + 8)     // LDS row stride (bf16 elems) for x tile, keeps 16B align
#define HC     128           // H chunk width
#define HS     (HC + 8)      // LDS row stride for h tile

typedef __bf16 bf16_t;
typedef bf16_t v16bf __attribute__((ext_vector_type(16)));
typedef float  v8f   __attribute__((ext_vector_type(8)));
typedef unsigned short u16;

union Frag16 { v16bf v; uint4 u4[2]; };

__device__ __forceinline__ u16 f32_to_bf16_rne(float f) {
    unsigned u = __float_as_uint(f);
    u += 0x7FFFu + ((u >> 16) & 1u);       // round-to-nearest-even
    return (u16)(u >> 16);
}

// ---------------------------------------------------------------------------
// Zero output + expert counters
// ---------------------------------------------------------------------------
__global__ void zero_init_kernel(float4* out4, int n4, int* ecnt) {
    int i = blockIdx.x * blockDim.x + threadIdx.x;
    if (i < n4) out4[i] = make_float4(0.f, 0.f, 0.f, 0.f);
    if (i < NEXP) ecnt[i] = 0;
}

// ---------------------------------------------------------------------------
// [R][C] f32  ->  [C][R] bf16  (per expert in blockIdx.z), LDS-tiled
// ---------------------------------------------------------------------------
__global__ __launch_bounds__(256) void transpose_cvt_kernel(const float* __restrict__ src,
                                                            u16* __restrict__ dst,
                                                            int R, int C) {
    __shared__ float tile[32][33];
    const size_t per = (size_t)R * C;
    src += per * blockIdx.z;
    dst += per * blockIdx.z;
    const int x0 = blockIdx.x * 32, y0 = blockIdx.y * 32;
    const int tx = threadIdx.x, ty = threadIdx.y;           // (32, 8)
#pragma unroll
    for (int i = 0; i < 4; ++i)
        tile[ty + i * 8][tx] = src[(size_t)(y0 + ty + i * 8) * C + x0 + tx];
    __syncthreads();
#pragma unroll
    for (int i = 0; i < 4; ++i)
        dst[(size_t)(x0 + ty + i * 8) * R + y0 + tx] = f32_to_bf16_rne(tile[tx][ty + i * 8]);
}

// ---------------------------------------------------------------------------
// x f32 -> bf16, coalesced
// ---------------------------------------------------------------------------
__global__ void cvt_x_kernel(const float4* __restrict__ x4, ushort4* __restrict__ xb4, int n4) {
    int i = blockIdx.x * blockDim.x + threadIdx.x;
    if (i >= n4) return;
    float4 v = x4[i];
    ushort4 o;
    o.x = f32_to_bf16_rne(v.x); o.y = f32_to_bf16_rne(v.y);
    o.z = f32_to_bf16_rne(v.z); o.w = f32_to_bf16_rne(v.w);
    xb4[i] = o;
}

// ---------------------------------------------------------------------------
// Gate: sigmoid(x @ Wg^T), top-2, normalized weights, append to expert lists.
// One wave32 per token.
// ---------------------------------------------------------------------------
__global__ __launch_bounds__(256) void gate_kernel(const float* __restrict__ x,
                                                   const float* __restrict__ Wg,
                                                   int* __restrict__ etok,
                                                   float* __restrict__ ewt,
                                                   int* __restrict__ ecnt) {
    const int lane  = threadIdx.x & 31;
    const int token = blockIdx.x * 8 + (threadIdx.x >> 5);
    if (token >= NTOK) return;

    const float* xr = x + (size_t)token * DIM;
    float acc[NEXP];
#pragma unroll
    for (int e = 0; e < NEXP; ++e) acc[e] = 0.f;

    for (int k = lane; k < DIM; k += 32) {
        float xv = xr[k];
#pragma unroll
        for (int e = 0; e < NEXP; ++e) acc[e] += xv * Wg[e * DIM + k];
    }
#pragma unroll
    for (int e = 0; e < NEXP; ++e) {
#pragma unroll
        for (int off = 16; off > 0; off >>= 1)
            acc[e] += __shfl_xor(acc[e], off, 32);
    }
    if (lane == 0) {
        float s[NEXP];
#pragma unroll
        for (int e = 0; e < NEXP; ++e) s[e] = 1.f / (1.f + __expf(-acc[e]));
        int b0 = 0;
#pragma unroll
        for (int e = 1; e < NEXP; ++e) if (s[e] > s[b0]) b0 = e;
        int b1 = (b0 == 0) ? 1 : 0;
#pragma unroll
        for (int e = 0; e < NEXP; ++e) if (e != b0 && s[e] > s[b1]) b1 = e;
        float inv = 1.f / (s[b0] + s[b1]);        // ROUTE_SCALE == 1
        float w0 = s[b0] * inv, w1 = s[b1] * inv;
        int s0 = atomicAdd(&ecnt[b0], 1);
        etok[b0 * NTOK + s0] = token; ewt[b0 * NTOK + s0] = w0;
        int s1 = atomicAdd(&ecnt[b1], 1);
        etok[b1 * NTOK + s1] = token; ewt[b1 * NTOK + s1] = w1;
    }
}

// ---------------------------------------------------------------------------
// Fused expert MLP: out[t] += w * ( relu(x[t] @ W1[e] + b1[e]) @ W2[e] + b2[e] )
// Tile = 32 tokens of one expert; 8 waves; wave w owns output cols [128w,128w+128)
// as 2 row-tiles x 8 col-tiles of persistent f32 WMMA accumulators.
// W1T: [E][H][D] bf16 (K-major), W2T: [E][D][H] bf16 (K-major).
// Each B-fragment (one contiguous 32B load per lane) feeds 2 WMMAs.
// ---------------------------------------------------------------------------
__global__ __launch_bounds__(256, 1) void ffn_kernel(const u16* __restrict__ xbf,
                                                     const u16* __restrict__ w1t,
                                                     const u16* __restrict__ w2t,
                                                     const float* __restrict__ b1,
                                                     const float* __restrict__ b2,
                                                     const int* __restrict__ etok,
                                                     const float* __restrict__ ewt,
                                                     const int* __restrict__ ecnt,
                                                     float* __restrict__ out) {
    const int e    = blockIdx.y;
    const int cnt  = ecnt[e];
    const int base = blockIdx.x * MTOK;
    if (base >= cnt) return;

    __shared__ u16   xlds[MTOK * XS];
    __shared__ u16   hlds[MTOK * HS];
    __shared__ int   tokS[MTOK];
    __shared__ float wtS[MTOK];

    const int tid = threadIdx.x;
    if (tid < MTOK) {
        int s = base + tid;
        if (s < cnt) { tokS[tid] = etok[e * NTOK + s]; wtS[tid] = ewt[e * NTOK + s]; }
        else         { tokS[tid] = 0;                  wtS[tid] = 0.f; }
    }
    __syncthreads();

    // Stage 32x1024 bf16 x-tile (gather by token id), 8 elems (uint4) per step.
    for (int i = tid; i < MTOK * (DIM / 8); i += 256) {
        int row = i >> 7, c8 = i & 127;
        uint4 v = *(const uint4*)(xbf + (size_t)tokS[row] * DIM + c8 * 8);
        *(uint4*)(&xlds[row * XS + c8 * 8]) = v;
    }
    __syncthreads();

    const int wave = tid >> 5, lane = tid & 31;
    const int half = lane >> 4, l15 = lane & 15;

    // acc[j]   : rows  0..15, cols wave*128 + j*16 ..
    // acc[8+j] : rows 16..31, cols wave*128 + j*16 ..
    v8f acc[16];
#pragma unroll
    for (int j = 0; j < 16; ++j)
#pragma unroll
        for (int r = 0; r < 8; ++r) acc[j][r] = 0.f;

    const u16*   w1e = w1t + (size_t)e * INTER * DIM;   // [H][D]
    const u16*   w2e = w2t + (size_t)e * DIM * INTER;   // [D][H]
    const float* b1e = b1 + (size_t)e * INTER;
    const float* b2e = b2 + (size_t)e * DIM;

    for (int hc = 0; hc < INTER / HC; ++hc) {
        const int hbase = hc * HC;

        // ---- layer 1: this wave computes h cols [hbase+16w, +16) for 32 rows ----
        const int  bcol  = hbase + wave * 16 + l15;           // W1T row (= h column)
        const u16* w1col = w1e + (size_t)bcol * DIM + half * 16;
        v8f hacc0, hacc1;
#pragma unroll
        for (int r = 0; r < 8; ++r) { hacc0[r] = 0.f; hacc1[r] = 0.f; }

#pragma unroll 4
        for (int k0 = 0; k0 < DIM; k0 += 32) {
            Frag16 a0, a1;
            a0.u4[0] = *(const uint4*)(&xlds[l15 * XS + k0 + half * 16]);
            a0.u4[1] = *(const uint4*)(&xlds[l15 * XS + k0 + half * 16 + 8]);
            a1.u4[0] = *(const uint4*)(&xlds[(16 + l15) * XS + k0 + half * 16]);
            a1.u4[1] = *(const uint4*)(&xlds[(16 + l15) * XS + k0 + half * 16 + 8]);
            v16bf b = *(const v16bf*)(w1col + k0);
            hacc0 = __builtin_amdgcn_wmma_f32_16x16x32_bf16(false, a0.v, false, b,
                                                            (short)0, hacc0, false, false);
            hacc1 = __builtin_amdgcn_wmma_f32_16x16x32_bf16(false, a1.v, false, b,
                                                            (short)0, hacc1, false, false);
        }
        // prefetch next H-chunk of W1T while layer 2 runs
        if (hc + 1 < INTER / HC)
            __builtin_prefetch(w1col + HC * DIM, 0, 1);

        __syncthreads();   // previous chunk's layer-2 reads of hlds are done
        {
            float bv = b1e[bcol];
#pragma unroll
            for (int r = 0; r < 8; ++r) {
                float v0 = hacc0[r] + bv;
                float v1 = hacc1[r] + bv;
                v0 = v0 > 0.f ? v0 : 0.f;                      // ReLU
                v1 = v1 > 0.f ? v1 : 0.f;
                hlds[(r + half * 8) * HS + wave * 16 + l15]        = f32_to_bf16_rne(v0);
                hlds[(16 + r + half * 8) * HS + wave * 16 + l15]   = f32_to_bf16_rne(v1);
            }
        }
        __syncthreads();

        // ---- layer 2: acc[32 x 128 slice] += h[32xHC] @ W2T chunk ----
        for (int kh = 0; kh < HC; kh += 32) {
            Frag16 a0, a1;
            a0.u4[0] = *(const uint4*)(&hlds[l15 * HS + kh + half * 16]);
            a0.u4[1] = *(const uint4*)(&hlds[l15 * HS + kh + half * 16 + 8]);
            a1.u4[0] = *(const uint4*)(&hlds[(16 + l15) * HS + kh + half * 16]);
            a1.u4[1] = *(const uint4*)(&hlds[(16 + l15) * HS + kh + half * 16 + 8]);
#pragma unroll
            for (int j = 0; j < 8; ++j) {
                const int  dcol = wave * 128 + j * 16 + l15;   // W2T row (= out column)
                v16bf b = *(const v16bf*)(w2e + (size_t)dcol * INTER + hbase + kh + half * 16);
                acc[j]     = __builtin_amdgcn_wmma_f32_16x16x32_bf16(false, a0.v, false, b,
                                                                     (short)0, acc[j], false, false);
                acc[8 + j] = __builtin_amdgcn_wmma_f32_16x16x32_bf16(false, a1.v, false, b,
                                                                     (short)0, acc[8 + j], false, false);
            }
        }
        __syncthreads();
    }

    // ---- epilogue: out[tok][d] += w_tok * (acc + b2) ----
#pragma unroll
    for (int j = 0; j < 8; ++j) {
        const int   dcol = wave * 128 + j * 16 + l15;
        const float b2v  = b2e[dcol];
#pragma unroll
        for (int r = 0; r < 8; ++r) {
            const int row0 = r + half * 8;
            const int row1 = 16 + row0;
            const float wt0 = wtS[row0];
            const float wt1 = wtS[row1];
            if (wt0 != 0.f)
                atomicAdd(&out[(size_t)tokS[row0] * DIM + dcol], (acc[j][r] + b2v) * wt0);
            if (wt1 != 0.f)
                atomicAdd(&out[(size_t)tokS[row1] * DIM + dcol], (acc[8 + j][r] + b2v) * wt1);
        }
    }
}

// ---------------------------------------------------------------------------
// Launcher. Inputs: x, Wg, W1, b1, W2, b2 (all f32). Output: y [T, D] f32.
// ---------------------------------------------------------------------------
extern "C" void kernel_launch(void* const* d_in, const int* in_sizes, int n_in,
                              void* d_out, int out_size, void* d_ws, size_t ws_size,
                              hipStream_t stream) {
    const float* x  = (const float*)d_in[0];
    const float* Wg = (const float*)d_in[1];
    const float* W1 = (const float*)d_in[2];
    const float* b1 = (const float*)d_in[3];
    const float* W2 = (const float*)d_in[4];
    const float* b2 = (const float*)d_in[5];
    float* out = (float*)d_out;

    // Workspace layout (bytes); total ~84.5 MB.
    char* ws = (char*)d_ws;
    const size_t W1T_OFF  = 0;
    const size_t W1T_SZ   = (size_t)NEXP * INTER * DIM * 2;      // 33.6 MB
    const size_t W2T_OFF  = W1T_OFF + W1T_SZ;
    const size_t W2T_SZ   = (size_t)NEXP * DIM * INTER * 2;      // 33.6 MB
    const size_t XBF_OFF  = W2T_OFF + W2T_SZ;
    const size_t XBF_SZ   = (size_t)NTOK * DIM * 2;              // 16.8 MB
    const size_t ETOK_OFF = XBF_OFF + XBF_SZ;
    const size_t ETOK_SZ  = (size_t)NEXP * NTOK * 4;
    const size_t EWT_OFF  = ETOK_OFF + ETOK_SZ;
    const size_t EWT_SZ   = (size_t)NEXP * NTOK * 4;
    const size_t ECNT_OFF = EWT_OFF + EWT_SZ;

    u16*   w1t  = (u16*)(ws + W1T_OFF);
    u16*   w2t  = (u16*)(ws + W2T_OFF);
    u16*   xbf  = (u16*)(ws + XBF_OFF);
    int*   etok = (int*)(ws + ETOK_OFF);
    float* ewt  = (float*)(ws + EWT_OFF);
    int*   ecnt = (int*)(ws + ECNT_OFF);

    // 1) zero output + counters
    {
        int n4 = out_size / 4;
        zero_init_kernel<<<(n4 + 255) / 256, 256, 0, stream>>>((float4*)out, n4, ecnt);
    }
    // 2) weight conversion/transpose to bf16 K-major
    {
        dim3 blk(32, 8, 1);
        dim3 g1(INTER / 32, DIM / 32, NEXP);     // W1 [D][H] -> W1T [H][D]
        transpose_cvt_kernel<<<g1, blk, 0, stream>>>(W1, w1t, DIM, INTER);
        dim3 g2(DIM / 32, INTER / 32, NEXP);     // W2 [H][D] -> W2T [D][H]
        transpose_cvt_kernel<<<g2, blk, 0, stream>>>(W2, w2t, INTER, DIM);
    }
    // 3) x -> bf16
    {
        int n4 = (NTOK * DIM) / 4;
        cvt_x_kernel<<<(n4 + 255) / 256, 256, 0, stream>>>((const float4*)x, (ushort4*)xbf, n4);
    }
    // 4) gating + routing
    gate_kernel<<<NTOK / 8, 256, 0, stream>>>(x, Wg, etok, ewt, ecnt);
    // 5) fused expert FFN (worst case: one expert owns all 8192 tokens -> 256 tiles)
    {
        dim3 grid(NTOK / MTOK, NEXP, 1);
        ffn_kernel<<<grid, 256, 0, stream>>>(xbf, w1t, w2t, b1, b2, etok, ewt, ecnt, out);
    }
}
